// SLSTM_56324201120045
// MI455X (gfx1250) — matrix operand order repaired and checked by reading, not compiled, not run on hardware
//
#include <hip/hip_runtime.h>
#include <hip/hip_bf16.h>
#include <math.h>

// ---------------------------------------------------------------------------
// sLSTM forward for MI455X (gfx1250, wave32, WMMA).
// gates = x_t @ Wih^T + h @ Whh^T + b  computed with v_wmma_f32_16x16x32_bf16,
// persistent-grid recurrence with one device-wide barrier per timestep.
// ---------------------------------------------------------------------------

typedef __bf16 bf16_t;
typedef __attribute__((ext_vector_type(16))) __bf16 v16bf;
typedef __attribute__((ext_vector_type(8)))  __bf16 v8bf;
typedef __attribute__((ext_vector_type(4)))  __bf16 v4bf;
typedef __attribute__((ext_vector_type(8)))  float  v8f;

#define SEQ   512
#define BATCH 64
#define DIN   1024
#define HID   1024
#define NG    4096            // 4 * HID gate columns
#define KTOT  64              // 2048 / 32 k-tiles (x part: 0..31, h part: 32..63)
#define NWG   64              // persistent workgroups, each owns 16 hidden units
#define GS    65              // LDS row stride for the 64x64 gate slab (pad)

// --------------------------- prep kernels ----------------------------------

// Pack W^T (concat of Wih,Whh along K) in bf16 into the exact WMMA B-fragment
// layout: block (gt,kt) is 32 lanes x 16 bf16 contiguous; lane l supplies
// column g = gt*16 + (l&15), K = kt*32 + (l>=16)*8 + (j&7) + (j>>3)*16  (j = 0..15).
__global__ __launch_bounds__(256) void slstm_pack_w(
    const float* __restrict__ wih, const float* __restrict__ whh,
    bf16_t* __restrict__ wpack)
{
  const int blk = blockIdx.x;          // 0 .. 256*64-1  (gt*64 + kt)
  const int gt  = blk >> 6;
  const int kt  = blk & 63;
  #pragma unroll
  for (int u = 0; u < 2; ++u) {
    const int e    = threadIdx.x * 2 + u;   // 0..511 within block
    const int lane = e >> 4;
    const int j    = e & 15;
    const int g    = gt * 16 + (lane & 15);
    const int k    = kt * 32 + ((lane >> 4) << 3) + (j & 7) + ((j >> 3) << 4);   // 0..2047
    const float w  = (k < DIN) ? wih[(size_t)g * DIN + k]
                               : whh[(size_t)g * HID + (k - DIN)];
    wpack[(size_t)blk * 512 + e] = (bf16_t)w;
  }
}

// Convert x (f32, row-major [SEQ][BATCH][DIN]) to bf16 once.
__global__ __launch_bounds__(256) void slstm_cvt_x(
    const float* __restrict__ x, bf16_t* __restrict__ xbf, int n)
{
  const int i = (blockIdx.x * 256 + threadIdx.x) * 4;
  if (i < n) {
    const float4 v = *(const float4*)(x + i);
    v4bf o;
    o.x = (__bf16)v.x; o.y = (__bf16)v.y; o.z = (__bf16)v.z; o.w = (__bf16)v.w;
    *(v4bf*)(xbf + i) = o;
  }
}

// Zero both h double-buffers (h0 = 0) and the grid-barrier counter.
__global__ __launch_bounds__(256) void slstm_init(
    bf16_t* __restrict__ hbf, unsigned* __restrict__ counter)
{
  const int i = blockIdx.x * 256 + threadIdx.x;
  if (i < 2 * BATCH * HID) hbf[i] = (bf16_t)0.0f;
  if (i == 0) *counter = 0u;
}

// --------------------------- main recurrence --------------------------------

__global__ __launch_bounds__(256) void slstm_persistent(
    const bf16_t* __restrict__ xbf,
    const bf16_t* __restrict__ wpack,
    bf16_t*       __restrict__ hbf,       // 2 buffers of BATCH*HID bf16
    const float*  __restrict__ bias_ih,
    const float*  __restrict__ bias_hh,
    float*        __restrict__ out,       // SEQ*BATCH*HID, then h_f, then c_f
    unsigned*     __restrict__ counter)
{
  const int wg    = blockIdx.x;           // owns hidden units [wg*16, wg*16+16)
  const int tid   = threadIdx.x;
  const int wave  = tid >> 5;
  const int lane  = tid & 31;
  const int lhalf = lane >> 4;            // A: lanes>=16 take K runs {8..15,24..31}
  const int l16   = lane & 15;

  // Wave w computes C tiles (q0,mt) and (q1,mt): same batch rows, two gates.
  const int q0 = wave >> 2;               // gate quadrant 0/1
  const int q1 = q0 + 2;                  // gate quadrant 2/3
  const int mt = wave & 3;                // batch tile (rows mt*16..mt*16+15)
  const int arow = mt * 16 + l16;         // A-matrix row this lane supplies

  const size_t gt0 = (size_t)(q0 * 64 + wg);   // 16-col group index in [0,256)
  const size_t gt1 = (size_t)(q1 * 64 + wg);

  __shared__ float gsm[64 * GS];          // gate slab: rows=batch, cols=4x16
  __shared__ float msm[64 * 16];          // m_new (for roll along batch)

  // Elementwise ownership: thread -> (b = tid/4, jj = (tid%4)*4 + kk)
  const int eb  = tid >> 2;
  const int ej0 = (tid & 3) * 4;

  float bsum[4][4];
  #pragma unroll
  for (int q = 0; q < 4; ++q)
    #pragma unroll
    for (int kk = 0; kk < 4; ++kk) {
      const int col = q * HID + wg * 16 + ej0 + kk;
      bsum[q][kk] = bias_ih[col] + bias_hh[col];
    }

  float cst[4] = {0.f, 0.f, 0.f, 0.f};
  float nst[4] = {1.f, 1.f, 1.f, 1.f};
  float mst[4] = {0.f, 0.f, 0.f, 0.f};
  float hlast[4] = {0.f, 0.f, 0.f, 0.f};

  for (int t = 0; t < SEQ; ++t) {
    const bf16_t* xt  = xbf + (size_t)t * (BATCH * DIN);
    const bf16_t* hrd = hbf + (size_t)(t & 1) * (BATCH * HID);

    v8f acc0 = {0.f, 0.f, 0.f, 0.f, 0.f, 0.f, 0.f, 0.f};
    v8f acc1 = {0.f, 0.f, 0.f, 0.f, 0.f, 0.f, 0.f, 0.f};

    // ---- x_t @ Wih^T contribution (k-tiles 0..31) ----
    {
      const bf16_t* abase = xt + (size_t)arow * DIN + lhalf * 8;
      #pragma unroll 4
      for (int kt = 0; kt < 32; ++kt) {
        const v8bf alo = *(const v8bf*)(abase + kt * 32);
        const v8bf ahi = *(const v8bf*)(abase + kt * 32 + 16);
        const v16bf a  = __builtin_shufflevector(alo, ahi,
            0, 1, 2, 3, 4, 5, 6, 7, 8, 9, 10, 11, 12, 13, 14, 15);
        const v16bf b0 = *(const v16bf*)(wpack + (gt0 * KTOT + kt) * 512 + lane * 16);
        const v16bf b1 = *(const v16bf*)(wpack + (gt1 * KTOT + kt) * 512 + lane * 16);
        acc0 = __builtin_amdgcn_wmma_f32_16x16x32_bf16(
                   false, a, false, b0, (short)0, acc0, false, false);
        acc1 = __builtin_amdgcn_wmma_f32_16x16x32_bf16(
                   false, a, false, b1, (short)0, acc1, false, false);
      }
    }
    // ---- h_{t-1} @ Whh^T contribution (k-tiles 32..63) ----
    {
      const bf16_t* abase = hrd + (size_t)arow * HID + lhalf * 8;
      #pragma unroll 4
      for (int kt = 0; kt < 32; ++kt) {
        const v8bf alo = *(const v8bf*)(abase + kt * 32);
        const v8bf ahi = *(const v8bf*)(abase + kt * 32 + 16);
        const v16bf a  = __builtin_shufflevector(alo, ahi,
            0, 1, 2, 3, 4, 5, 6, 7, 8, 9, 10, 11, 12, 13, 14, 15);
        const v16bf b0 = *(const v16bf*)(wpack + (gt0 * KTOT + 32 + kt) * 512 + lane * 16);
        const v16bf b1 = *(const v16bf*)(wpack + (gt1 * KTOT + 32 + kt) * 512 + lane * 16);
        acc0 = __builtin_amdgcn_wmma_f32_16x16x32_bf16(
                   false, a, false, b0, (short)0, acc0, false, false);
        acc1 = __builtin_amdgcn_wmma_f32_16x16x32_bf16(
                   false, a, false, b1, (short)0, acc1, false, false);
      }
    }

    // Deposit C tiles to LDS.  C layout: VGPR v -> row v (lanes 0-15) / v+8
    // (lanes 16-31); lane%16 -> column.
    #pragma unroll
    for (int v = 0; v < 8; ++v) {
      const int row = mt * 16 + v + lhalf * 8;
      gsm[row * GS + q0 * 16 + l16] = acc0[v];
      gsm[row * GS + q1 * 16 + l16] = acc1[v];
    }
    __syncthreads();

    // ---- elementwise sLSTM update (f32) ----
    float mnew[4], igs[4], fgs[4], zgs[4], ogs[4];
    #pragma unroll
    for (int kk = 0; kk < 4; ++kk) {
      const int jj = ej0 + kk;
      const float ig = gsm[eb * GS +       jj] + bsum[0][kk];
      const float fg = gsm[eb * GS + 16 +  jj] + bsum[1][kk];
      const float zg = gsm[eb * GS + 32 +  jj] + bsum[2][kk];
      const float og = gsm[eb * GS + 48 +  jj] + bsum[3][kk];
      const float mn = fmaxf(fg + mst[kk], ig);
      msm[eb * 16 + jj] = mn;
      mnew[kk] = mn; igs[kk] = ig; fgs[kk] = fg; zgs[kk] = zg; ogs[kk] = og;
    }
    __syncthreads();

    bf16_t* hwr = hbf + (size_t)((t + 1) & 1) * (BATCH * HID);
    #pragma unroll
    for (int kk = 0; kk < 4; ++kk) {
      const int jj = ej0 + kk;
      const float mprev = msm[((eb + 63) & 63) * 16 + jj];   // roll(m_new,1,batch)
      const float ip = __expf(igs[kk] - mnew[kk]);
      const float fp = __expf(fgs[kk] + mprev - mnew[kk]);
      const float cn = fp * cst[kk] + ip * tanhf(zgs[kk]);
      const float nn = fp * nst[kk] + ip;
      const float hn = (cn / nn) / (1.f + __expf(-ogs[kk]));
      cst[kk] = cn; nst[kk] = nn; mst[kk] = mnew[kk]; hlast[kk] = hn;
      out[(size_t)t * (BATCH * HID) + (size_t)eb * HID + wg * 16 + jj] = hn;
      hwr[eb * HID + wg * 16 + jj] = (bf16_t)hn;
    }

    // ---- device-wide barrier: h(t) visible to all WGs before step t+1 ----
    __syncthreads();
    __threadfence();
    if (tid == 0) {
      __hip_atomic_fetch_add(counter, 1u, __ATOMIC_RELEASE,
                             __HIP_MEMORY_SCOPE_AGENT);
      const unsigned target = (unsigned)NWG * (unsigned)(t + 1);
      while (__hip_atomic_load(counter, __ATOMIC_ACQUIRE,
                               __HIP_MEMORY_SCOPE_AGENT) < target)
        __builtin_amdgcn_s_sleep(2);
    }
    __syncthreads();
    __threadfence();
  }

  // ---- final states (h_f, c_f) appended after the [SEQ,BATCH,HID] output ----
  float* hf = out + (size_t)SEQ * BATCH * HID;
  float* cf = hf + BATCH * HID;
  #pragma unroll
  for (int kk = 0; kk < 4; ++kk) {
    const int jj = ej0 + kk;
    hf[eb * HID + wg * 16 + jj] = hlast[kk];
    cf[eb * HID + wg * 16 + jj] = cst[kk];
  }
}

// --------------------------- launch ----------------------------------------

extern "C" void kernel_launch(void* const* d_in, const int* in_sizes, int n_in,
                              void* d_out, int out_size, void* d_ws, size_t ws_size,
                              hipStream_t stream) {
  const float* x   = (const float*)d_in[0];   // [SEQ, BATCH, DIN]
  const float* wih = (const float*)d_in[1];   // [4H, DIN]
  const float* whh = (const float*)d_in[2];   // [4H, H]
  const float* bih = (const float*)d_in[3];   // [4H]
  const float* bhh = (const float*)d_in[4];   // [4H]
  float* out = (float*)d_out;

  // Workspace layout (~81 MiB):
  //   [0, 16 MiB)      bf16 packed weights (B-fragment layout, K=2048 x N=4096)
  //   [16, 80 MiB)     bf16 copy of x
  //   [80 MiB, +256KB) h double buffer (bf16)
  //   [+1 MiB]         grid-barrier counter
  char* ws = (char*)d_ws;
  bf16_t*   wpack   = (bf16_t*)ws;
  bf16_t*   xbf     = (bf16_t*)(ws + ((size_t)16 << 20));
  bf16_t*   hbf     = (bf16_t*)(ws + ((size_t)80 << 20));
  unsigned* counter = (unsigned*)(ws + ((size_t)80 << 20) + ((size_t)1 << 20));

  slstm_pack_w<<<256 * 64, 256, 0, stream>>>(wih, whh, wpack);
  slstm_cvt_x<<<(SEQ * BATCH * DIN) / (4 * 256), 256, 0, stream>>>(
      x, xbf, SEQ * BATCH * DIN);
  slstm_init<<<(2 * BATCH * HID + 255) / 256, 256, 0, stream>>>(hbf, counter);
  slstm_persistent<<<NWG, 256, 0, stream>>>(xbf, wpack, hbf, bih, bhh, out,
                                            counter);
}
